// CSPNFusion_6468220748301
// MI455X (gfx1250) — compile-verified, matrix operand order checked by reading, not run on hardware
//
#include <hip/hip_runtime.h>
#include <hip/hip_bf16.h>

#define Nn 2
#define Vv 4
#define Ff 64
#define Cc 3
#define Hh 384
#define Ww 384
#define HW (Hh*Ww)
#define NV (Nn*Vv)
#define NREPS 6
#define TILES_X (Ww/16)          // 24
#define TILES_TOTAL (NV*Hh*TILES_X) // 73728
#define WPB 4                    // waves per block

typedef __attribute__((ext_vector_type(16))) __bf16 v16bf;
typedef __attribute__((ext_vector_type(8)))  __bf16 v8bf;
typedef __attribute__((ext_vector_type(8)))  float  v8f;

__device__ __forceinline__ __bf16 f2bf(float f) {
    unsigned u = __builtin_bit_cast(unsigned, f);
    unsigned r = u + 0x7FFFu + ((u >> 16) & 1u);   // round-to-nearest-even
    unsigned short h = (unsigned short)(r >> 16);
    return __builtin_bit_cast(__bf16, h);
}

__device__ __forceinline__ float sigmoidf(float z) {
    return 1.0f / (1.0f + __expf(-z));
}

// ---------------------------------------------------------------------------
// Kernel 0: pack combined conv weights into WMMA B-fragment layout.
// B[k = tap*64 + f][n]:  n<9 guide*s, n==9 gview*s, n==10 gating*s, else 0.
// Fragment storage: frag[t][c][lane][e], lane holds col n=lane&15,
// K = kb+{0..7} then kb+16+{0..7}, kb = (lane>>4)*8, f = c*32 + K.
// ---------------------------------------------------------------------------
__global__ void pack_weights_k(const float* __restrict__ guide_w,
                               const float* __restrict__ guide_s,
                               const float* __restrict__ gview_w,
                               const float* __restrict__ gview_s,
                               const float* __restrict__ gating_w,
                               const float* __restrict__ gating_s,
                               __bf16* __restrict__ Bpack) {
    int i = blockIdx.x * blockDim.x + threadIdx.x;
    if (i >= 9 * 2 * 32 * 16) return;
    int e    = i & 15;
    int lane = (i >> 4) & 31;
    int c    = (i >> 9) & 1;
    int t    = i >> 10;
    int n    = lane & 15;
    int kb   = (lane >> 4) * 8;
    int kk   = (e < 8) ? (kb + e) : (kb + 16 + (e - 8));
    int f    = c * 32 + kk;
    float val = 0.0f;
    if (n < 9)        val = guide_w[(n * Ff + f) * 9 + t] * guide_s[n];
    else if (n == 9)  val = gview_w[f * 9 + t] * gview_s[0];
    else if (n == 10) val = gating_w[f * 9 + t] * gating_s[0];
    Bpack[i] = f2bf(val);
}

// ---------------------------------------------------------------------------
// Kernel 1: conv-as-GEMM via v_wmma_f32_16x16x32_bf16.
// One wave = one 16-pixel row tile of one view image. K = 9 taps * 64 ch.
// Epilogue: softmax over 9 guide cols, sigmoid(gview)*mask, raw gating col.
// ---------------------------------------------------------------------------
__global__ __launch_bounds__(32 * WPB)
void conv_wmma_k(const float* __restrict__ gf,       // [NV,F,H,W]
                 const float* __restrict__ masks,    // [NV,1,H,W]
                 const __bf16* __restrict__ Bpack,   // [9][2][32][16]
                 const float* __restrict__ guide_b,  // [9]
                 const float* __restrict__ gview_b,  // [1]
                 float* __restrict__ guide_ws,       // [NV,9,H,W] softmaxed
                 float* __restrict__ gview_ws,       // [NV,H,W] sig*mask
                 float* __restrict__ gatp_ws) {      // [NV,H,W] raw gating col
    __shared__ __bf16 sB[9 * 2 * 32 * 16];            // 18 KB
    __shared__ __bf16 sA[WPB][3 * 18 * Ff];           // 4 * 6912 B
    __shared__ float  sO[WPB][16 * 16];               // 4 KB

    const int tid  = threadIdx.x;
    const int lane = tid & 31;
    const int wv   = tid >> 5;

    // stage B fragments (whole block cooperatively)
    {
        const unsigned* src = (const unsigned*)Bpack;
        unsigned* dst = (unsigned*)sB;
        for (int j = tid; j < (9 * 2 * 32 * 16) / 2; j += 32 * WPB) dst[j] = src[j];
    }

    const int tile = blockIdx.x * WPB + wv;
    const int img  = tile / (Hh * TILES_X);
    const int rem  = tile % (Hh * TILES_X);
    const int y0   = rem / TILES_X;
    const int x0   = (rem % TILES_X) * 16;

    // stage A halo tile: rows y0-1..y0+1, cols x0-1..x0+16, 64 ch, fp32->bf16
    // flat iteration index e2 = (r*64+f)*18 + xx  (xx fastest -> coalesced)
    const float* base = gf + (size_t)img * Ff * HW;
    for (int e2 = lane; e2 < 3 * Ff * 18; e2 += 32) {
        int xx = e2 % 18;
        int rf = e2 / 18;
        int f  = rf & 63;
        int r  = rf >> 6;
        int gy = y0 + r - 1;
        int gx = x0 + xx - 1;
        float v = 0.0f;
        if (gy >= 0 && gy < Hh && gx >= 0 && gx < Ww)
            v = base[(size_t)f * HW + gy * Ww + gx];
        sA[wv][(r * 18 + xx) * Ff + f] = f2bf(v);
    }
    __syncthreads();

    const int m    = lane & 15;
    const int kb   = (lane >> 4) * 8;
    v8f acc = {};

#pragma unroll
    for (int t = 0; t < 9; ++t) {
        const int dy = t / 3, dx = t % 3;
        const int p  = dy * 18 + m + dx;          // LDS pixel index (with halo)
#pragma unroll
        for (int c = 0; c < 2; ++c) {
            const __bf16* ap = &sA[wv][p * Ff + c * 32 + kb];
            v8bf alo = *(const v8bf*)ap;           // K = kb..kb+7
            v8bf ahi = *(const v8bf*)(ap + 16);    // K = kb+16..kb+23
            v16bf A  = __builtin_shufflevector(alo, ahi,
                        0,1,2,3,4,5,6,7,8,9,10,11,12,13,14,15);
            const __bf16* bp = &sB[(((t * 2 + c) * 32) + lane) * 16];
            v8bf blo = *(const v8bf*)bp;
            v8bf bhi = *(const v8bf*)(bp + 8);
            v16bf B  = __builtin_shufflevector(blo, bhi,
                        0,1,2,3,4,5,6,7,8,9,10,11,12,13,14,15);
            acc = __builtin_amdgcn_wmma_f32_16x16x32_bf16(
                      false, A, false, B, (short)0, acc, false, false);
        }
    }

    // spill 16x16 f32 accumulator to LDS: VGPR j, lane -> M = j+8*(lane>>4), N = lane&15
#pragma unroll
    for (int j = 0; j < 8; ++j)
        sO[wv][(j + 8 * (lane >> 4)) * 16 + m] = acc[j];

    // epilogue: lanes 0..15 each own one pixel (LDS is in-order within a wave)
    if (lane < 16) {
        const int gx = x0 + lane;
        float v[16];
#pragma unroll
        for (int k = 0; k < 16; ++k) v[k] = sO[wv][lane * 16 + k];

        // softmax over 9 guide taps (add folded BN bias first)
        float mx = -3.4e38f;
        float g[9];
#pragma unroll
        for (int k = 0; k < 9; ++k) { g[k] = v[k] + guide_b[k]; mx = fmaxf(mx, g[k]); }
        float se = 0.0f;
#pragma unroll
        for (int k = 0; k < 9; ++k) { g[k] = __expf(g[k] - mx); se += g[k]; }
        float inv = 1.0f / se;
        const size_t pix = (size_t)y0 * Ww + gx;
#pragma unroll
        for (int k = 0; k < 9; ++k)
            guide_ws[((size_t)(img * 9 + k)) * HW + pix] = g[k] * inv;

        float mk = masks[(size_t)img * HW + pix];
        gview_ws[(size_t)img * HW + pix] = sigmoidf(v[9] + gview_b[0]) * mk;
        gatp_ws [(size_t)img * HW + pix] = v[10];   // pre-mean, pre-bias gating
    }
}

// ---------------------------------------------------------------------------
// Kernel 2: gating mean+sigmoid+mask, CSPN iterations, view fusion, blend.
// One thread per (n, y, x).
// ---------------------------------------------------------------------------
__global__ void fuse_k(const float* __restrict__ deep,     // [N,C,H,W]
                       const float* __restrict__ proj,     // [NV,C,H,W]
                       const float* __restrict__ masks,    // [NV,1,H,W]
                       const float* __restrict__ gating_b, // [1]
                       const float* __restrict__ guide_ws,
                       const float* __restrict__ gview_ws,
                       const float* __restrict__ gatp_ws,
                       float* __restrict__ out) {
    int idx = blockIdx.x * blockDim.x + threadIdx.x;
    if (idx >= Nn * HW) return;
    const int n = idx / HW;
    const int p = idx % HW;
    const int y = p / Ww;
    const int x = p % Ww;

    // gating: sigmoid(mean_V(conv) + b) gated by (sum masks > 0)
    float gp = 0.0f, msum = 0.0f;
#pragma unroll
    for (int vi = 0; vi < Vv; ++vi) {
        const int img = n * Vv + vi;
        gp   += gatp_ws[(size_t)img * HW + p];
        msum += masks  [(size_t)img * HW + p];
    }
    float gating = sigmoidf(gp * (1.0f / Vv) + gating_b[0]);
    if (!(msum > 0.0f)) gating = 0.0f;

    float fused[Cc] = {0.0f, 0.0f, 0.0f};
    float gvsum = 0.0f;

#pragma unroll
    for (int vi = 0; vi < Vv; ++vi) {
        const int img = n * Vv + vi;
        float g[9];
#pragma unroll
        for (int k = 0; k < 9; ++k)
            g[k] = guide_ws[((size_t)(img * 9 + k)) * HW + p];
        const float gc = g[4];
        const float gv = gview_ws[(size_t)img * HW + p];
        gvsum += gv;

#pragma unroll
        for (int c = 0; c < Cc; ++c) {
            const float* pc = proj + ((size_t)(img * Cc + c)) * HW;
            float basev = 0.0f;
#pragma unroll
            for (int k = 0; k < 9; ++k) {
                if (k == 4) continue;
                const int gy = y + k / 3 - 1;
                const int gx = x + k % 3 - 1;
                float s = 0.0f;
                if (gy >= 0 && gy < Hh && gx >= 0 && gx < Ww)
                    s = pc[gy * Ww + gx];
                basev += g[k] * s;
            }
            float modv = pc[p];
#pragma unroll
            for (int it = 0; it < NREPS; ++it) modv = fmaf(gc, modv, basev);
            fused[c] = fmaf(modv, gv, fused[c]);
        }
    }

    const float inv = 1.0f / (gvsum + 1e-7f);
#pragma unroll
    for (int c = 0; c < Cc; ++c) {
        const float f = fused[c] * inv;
        const float d = deep[((size_t)(n * Cc + c)) * HW + p];
        const size_t o = ((size_t)(n * Cc + c)) * HW + p;
        out[o]                            = (f * gating + d * (1.0f - gating)) * 2.0f - 1.0f;
        out[(size_t)Nn * Cc * HW + o]     = f * 2.0f - 1.0f;
    }
}

// ---------------------------------------------------------------------------
extern "C" void kernel_launch(void* const* d_in, const int* in_sizes, int n_in,
                              void* d_out, int out_size, void* d_ws, size_t ws_size,
                              hipStream_t stream) {
    const float* gf       = (const float*)d_in[0];
    const float* deep     = (const float*)d_in[1];
    const float* proj     = (const float*)d_in[2];
    const float* masks    = (const float*)d_in[3];
    const float* gating_w = (const float*)d_in[4];
    const float* gating_s = (const float*)d_in[5];
    const float* gating_b = (const float*)d_in[6];
    const float* gview_w  = (const float*)d_in[7];
    const float* gview_s  = (const float*)d_in[8];
    const float* gview_b  = (const float*)d_in[9];
    const float* guide_w  = (const float*)d_in[10];
    const float* guide_s  = (const float*)d_in[11];
    const float* guide_b  = (const float*)d_in[12];

    char* ws = (char*)d_ws;
    __bf16* Bpack    = (__bf16*)(ws);                          // 18432 B
    float*  guide_ws = (float*)(ws + 32768);                   // NV*9*HW*4
    float*  gview_ws = (float*)(ws + 32768 + (size_t)NV*9*HW*4);
    float*  gatp_ws  = (float*)(ws + 32768 + (size_t)NV*9*HW*4 + (size_t)NV*HW*4);
    float*  out      = (float*)d_out;

    pack_weights_k<<<(9 * 2 * 32 * 16 + 255) / 256, 256, 0, stream>>>(
        guide_w, guide_s, gview_w, gview_s, gating_w, gating_s, Bpack);

    conv_wmma_k<<<TILES_TOTAL / WPB, 32 * WPB, 0, stream>>>(
        gf, masks, Bpack, guide_b, gview_b, guide_ws, gview_ws, gatp_ws);

    fuse_k<<<(Nn * HW + 255) / 256, 256, 0, stream>>>(
        deep, proj, masks, gating_b, guide_ws, gview_ws, gatp_ws, out);
}
